// CostVolume_21586505630010
// MI455X (gfx1250) — matrix operand order repaired and checked by reading, not compiled, not run on hardware
//
#include <hip/hip_runtime.h>

#define BD  2
#define ND  4096
#define MD  4096
#define CD  128
#define KD  16
#define HD  512
#define RD  (BD*ND*KD)    /* 131072 */
#define R2D (BD*ND)       /* 8192   */

typedef __bf16 bf16_t;
typedef __attribute__((ext_vector_type(16))) __bf16 v16bf;
typedef __attribute__((ext_vector_type(8)))  float  v8f;

// ---------------- workspace layout (bytes) ----------------
#define OFF_IDX     (0u)                       // int[131072]            512KB
#define OFF_W       (1u<<20)                   // bf16 weights           768KB
#define OFF_XW      (2u<<20)                   // bf16[8192*128]         2MB
#define OFF_Q       (4u<<20)                   // bf16[131072*128]       32MB
#define OFF_GI      (36u<<20)                  // f32[8192*128]          4MB
#define OFF_II      (40u<<20)                  // f32[8192*128]          4MB
#define OFF_FEAT    (44u<<20)                  // f32[131072*128]        64MB
#define OFF_ENCLIN  (108u<<20)                 // f32[131072*128]        64MB
#define OFF_Y0      (172u<<20)                 // f32[131072*256]        128MB
#define OFF_Y1      (300u<<20)                 // f32[131072*128]        64MB
#define OFF_STATS   (364u<<20)                 // f32[1024]

// bf16 weight sub-offsets (elements)
#define WGP_O 0
#define WGI_O 16384
#define W1P_O 32768
#define W2P_O 98304
#define W1I_O 163840
#define W2I_O 229376
#define W20_O 294912
#define W21_O 360448

// ---------------- WMMA fragment helpers ----------------
// A fragment: 16x32 bf16, row-major source with leading dim lda (elems).
// lanes 0-15: row l, halves[0..7]=K k0+0..7, halves[8..15]=K k0+16..23
// lanes 16-31: row l, halves[0..7]=K k0+8..15, halves[8..15]=K k0+24..31
__device__ inline v16bf load_fragA(const bf16_t* A, int lda, int k0, int lane) {
  int l  = lane & 15;
  int ko = (lane < 16) ? 0 : 8;
  const bf16_t* p = A + l * lda + k0 + ko;
  union { uint4 u[2]; v16bf v; } c;
  c.u[0] = *reinterpret_cast<const uint4*>(p);
  c.u[1] = *reinterpret_cast<const uint4*>(p + 16);
  return c.v;
}

// B fragment: 32x16 bf16 (K x N). W row-major (Cout, Cin): column n of B == row (col0+n) of W.
// lanes 0-15: col n, K k0+0..7,16..23 ; lanes 16-31: col n, K k0+8..15,24..31
__device__ inline v16bf load_fragB(const bf16_t* W, int ldw, int col0, int k0, int lane) {
  int n  = lane & 15;
  int ko = (lane < 16) ? 0 : 8;
  const bf16_t* p = W + (size_t)(col0 + n) * ldw + k0 + ko;
  union { uint4 u[2]; v16bf v; } c;
  c.u[0] = *reinterpret_cast<const uint4*>(p);
  c.u[1] = *reinterpret_cast<const uint4*>(p + 16);
  return c.v;
}

// Single 16x16 output tile
__device__ inline v8f gemm16(const bf16_t* A, int lda, const bf16_t* W, int ldw,
                             int col0, int kdim, float bias_v, int lane) {
  v8f acc;
#pragma unroll
  for (int j = 0; j < 8; ++j) acc[j] = bias_v;
  for (int k0 = 0; k0 < kdim; k0 += 32) {
    v16bf a = load_fragA(A, lda, k0, lane);
    v16bf b = load_fragB(W, ldw, col0, k0, lane);
    acc = __builtin_amdgcn_wmma_f32_16x16x32_bf16(false, a, false, b, (short)0, acc,
                                                  false, false);
  }
  return acc;
}

// NT adjacent 16x16 output tiles sharing one A fragment per k-step (A reuse in regs)
template <int NT>
__device__ inline void gemmN(const bf16_t* A, int lda, const bf16_t* W, int ldw,
                             int col0, int kdim, const float* bias, int lane,
                             v8f acc[NT]) {
#pragma unroll
  for (int t = 0; t < NT; ++t) {
    float bv = bias[col0 + t * 16 + (lane & 15)];
#pragma unroll
    for (int j = 0; j < 8; ++j) acc[t][j] = bv;
  }
  for (int k0 = 0; k0 < kdim; k0 += 32) {
    v16bf a = load_fragA(A, lda, k0, lane);
#pragma unroll
    for (int t = 0; t < NT; ++t) {
      v16bf b = load_fragB(W, ldw, col0 + t * 16, k0, lane);
      acc[t] = __builtin_amdgcn_wmma_f32_16x16x32_bf16(false, a, false, b, (short)0,
                                                       acc[t], false, false);
    }
  }
}

// C/D layout: lanes 0-15: N=lane, M=j ; lanes 16-31: N=lane-16, M=8+j
__device__ inline void store_tile(float* out, int ld, int row0, int col0, int lane, v8f c) {
  int ncol = lane & 15;
  int mb   = (lane < 16) ? 0 : 8;
#pragma unroll
  for (int j = 0; j < 8; ++j) out[(size_t)(row0 + mb + j) * ld + col0 + ncol] = c[j];
}

__device__ inline unsigned long long shfl_xor_u64(unsigned long long v, int mask) {
  unsigned lo = (unsigned)v, hi = (unsigned)(v >> 32);
  lo = __shfl_xor(lo, mask, 32);
  hi = __shfl_xor(hi, mask, 32);
  return ((unsigned long long)hi << 32) | lo;
}

__device__ inline float sigmoidf_(float x) { return 1.0f / (1.0f + __expf(-x)); }

// ---------------- kernels ----------------
__global__ __launch_bounds__(256) void k_zero(float* p, int n) {
  int i = blockIdx.x * blockDim.x + threadIdx.x;
  if (i < n) p[i] = 0.0f;
}

__global__ __launch_bounds__(256) void k_cvt4(const float* __restrict__ src,
                                              bf16_t* __restrict__ dst, int n4) {
  int i = blockIdx.x * blockDim.x + threadIdx.x;
  if (i >= n4) return;
  float4 v = reinterpret_cast<const float4*>(src)[i];
  union { bf16_t h[4]; uint2 u; } o;
  o.h[0] = (bf16_t)v.x; o.h[1] = (bf16_t)v.y; o.h[2] = (bf16_t)v.z; o.h[3] = (bf16_t)v.w;
  reinterpret_cast<uint2*>(dst)[i] = o.u;
}

// ---- KNN: one wave per query point, f2_xyz staged in LDS ----
__global__ __launch_bounds__(256) void k_knn(const float* __restrict__ wxyz,
                                             const float* __restrict__ f2xyz,
                                             int* __restrict__ idx_out) {
  __shared__ float sx[MD], sy[MD], sz[MD];       // 48KB
  int b = blockIdx.y;
  const float* f2 = f2xyz + (size_t)b * MD * 3;
  for (int i = threadIdx.x; i < MD; i += 256) {
    sx[i] = f2[3 * i]; sy[i] = f2[3 * i + 1]; sz[i] = f2[3 * i + 2];
  }
  __syncthreads();
  int wave = threadIdx.x >> 5, lane = threadIdx.x & 31;
  int n = blockIdx.x * 8 + wave;
  const float* q = wxyz + ((size_t)b * ND + n) * 3;
  float qx = q[0], qy = q[1], qz = q[2];

  float bd[16]; int bi[16];
#pragma unroll
  for (int t = 0; t < 16; ++t) { bd[t] = 3.4e38f; bi[t] = 0; }
  float worst = 3.4e38f;
  for (int j = lane; j < MD; j += 32) {
    float dx = sx[j] - qx, dy = sy[j] - qy, dz = sz[j] - qz;
    float d = dx * dx + dy * dy + dz * dz;
    if (d < worst) {
      int wp = 0; float wv = bd[0];
#pragma unroll
      for (int t = 1; t < 16; ++t) if (bd[t] > wv) { wv = bd[t]; wp = t; }
#pragma unroll
      for (int t = 0; t < 16; ++t) if (t == wp) { bd[t] = d; bi[t] = j; }
      wv = bd[0];
#pragma unroll
      for (int t = 1; t < 16; ++t) wv = fmaxf(wv, bd[t]);
      worst = wv;
    }
  }
  // cooperative merge: 16 rounds of cross-lane min
  unsigned used = 0;
  for (int r = 0; r < 16; ++r) {
    float mv = 3.4e38f; int mp = 0, idxc = 0;
#pragma unroll
    for (int t = 0; t < 16; ++t) {
      bool u = (used >> t) & 1;
      float vv = u ? 3.4e38f : bd[t];
      if (vv < mv) { mv = vv; mp = t; idxc = bi[t]; }
    }
    unsigned long long key =
        (((unsigned long long)__float_as_uint(mv)) << 32) | (unsigned)lane;
#pragma unroll
    for (int s = 16; s >= 1; s >>= 1) {
      unsigned long long o = shfl_xor_u64(key, s);
      key = (o < key) ? o : key;
    }
    int wl = (int)(key & 31u);
    int widx = __shfl(idxc, wl, 32);
    if (lane == wl) used |= 1u << mp;
    if (lane == 0) idx_out[((size_t)b * ND + n) * KD + r] = widx;
  }
}

// ---- gather neighbor points -> bf16 (R x 128) ----
__global__ __launch_bounds__(256) void k_gather(const float* __restrict__ f2pts,
                                                const int* __restrict__ idx,
                                                bf16_t* __restrict__ Q) {
  int t = blockIdx.x * blockDim.x + threadIdx.x;   // RD*32 threads
  int r = t >> 5;
  int c4 = (t & 31) << 2;
  int b = r >> 16;                                  // N*K = 65536 rows per batch
  int m = idx[r];
  float4 v = *reinterpret_cast<const float4*>(f2pts + ((size_t)b * MD + m) * CD + c4);
  union { bf16_t h[4]; uint2 u; } o;
  o.h[0] = (bf16_t)v.x; o.h[1] = (bf16_t)v.y; o.h[2] = (bf16_t)v.z; o.h[3] = (bf16_t)v.w;
  *reinterpret_cast<uint2*>(Q + (size_t)r * CD + c4) = o.u;
}

// ---- i-branch (per (b,n)): gi = sigmoid(Wgi x), ii = W2i (W1i x + b1) + b2 ----
__global__ __launch_bounds__(256) void k_ibranch(const bf16_t* __restrict__ Xw,
                                                 const bf16_t* __restrict__ Wgi, const float* __restrict__ bgi,
                                                 const bf16_t* __restrict__ W1,  const float* __restrict__ b1,
                                                 const bf16_t* __restrict__ W2,  const float* __restrict__ b2,
                                                 float* __restrict__ gi_out, float* __restrict__ ii_out) {
  __shared__ __align__(16) bf16_t As[16 * CD];
  __shared__ __align__(16) bf16_t Hs[16 * HD];
  int row0 = blockIdx.x * 16;
  { // stage A tile
    const uint2* src = reinterpret_cast<const uint2*>(Xw + (size_t)row0 * CD);
    uint2* dst = reinterpret_cast<uint2*>(As);
    for (int i = threadIdx.x; i < 16 * CD / 4; i += 256) dst[i] = src[i];
  }
  __syncthreads();
  int wave = threadIdx.x >> 5, lane = threadIdx.x & 31;
  int ncol = lane & 15, mb = (lane < 16) ? 0 : 8;
  {                                                // hidden: wave covers 64 cols
    v8f acc[4];
    int col0 = wave * 64;
    gemmN<4>(As, CD, W1, CD, col0, CD, b1, lane, acc);
#pragma unroll
    for (int t = 0; t < 4; ++t)
#pragma unroll
      for (int j = 0; j < 8; ++j)
        Hs[(mb + j) * HD + col0 + t * 16 + ncol] = (bf16_t)acc[t][j];
  }
  __syncthreads();
  {                                                // ii: 16x128
    int col0 = wave * 16;
    v8f acc = gemm16(Hs, HD, W2, HD, col0, HD, b2[col0 + ncol], lane);
    store_tile(ii_out, CD, row0, col0, lane, acc);
  }
  {                                                // gi: 16x128 + sigmoid
    int col0 = wave * 16;
    v8f acc = gemm16(As, CD, Wgi, CD, col0, CD, bgi[col0 + ncol], lane);
#pragma unroll
    for (int j = 0; j < 8; ++j) acc[j] = sigmoidf_(acc[j]);
    store_tile(gi_out, CD, row0, col0, lane, acc);
  }
}

// ---- main fused: p-branch GEMMs + gating + feat + enc linear (+stats0) ----
__global__ __launch_bounds__(256) void k_main(const bf16_t* __restrict__ Q,
                                              const bf16_t* __restrict__ Wgp, const float* __restrict__ bgp,
                                              const bf16_t* __restrict__ W1,  const float* __restrict__ b1,
                                              const bf16_t* __restrict__ W2,  const float* __restrict__ b2,
                                              const float* __restrict__ gi,   const float* __restrict__ ii,
                                              const float* __restrict__ wxyz, const float* __restrict__ f2xyz,
                                              const int* __restrict__ idx,
                                              const float* __restrict__ encW, const float* __restrict__ encB,
                                              float* __restrict__ feat_out, float* __restrict__ enclin_out,
                                              float* __restrict__ stats0) {
  __shared__ __align__(16) bf16_t As[16 * CD];
  __shared__ __align__(16) bf16_t Hs[16 * HD];
  __shared__ float xc[16][10];
  int bn = blockIdx.x;
  int row0 = bn * 16;
  { // stage A tile (the 16 neighbors of this (b,n))
    const uint2* src = reinterpret_cast<const uint2*>(Q + (size_t)row0 * CD);
    uint2* dst = reinterpret_cast<uint2*>(As);
    for (int i = threadIdx.x; i < 16 * CD / 4; i += 256) dst[i] = src[i];
  }
  if (threadIdx.x < 16) {   // xyz_cat rows
    int k = threadIdx.x;
    int b = bn >> 12;
    const float* p = wxyz + (size_t)bn * 3;
    float px = p[0], py = p[1], pz = p[2];
    int m = idx[row0 + k];
    const float* q = f2xyz + ((size_t)b * MD + m) * 3;
    float qx = q[0], qy = q[1], qz = q[2];
    float dx = qx - px, dy = qy - py, dz = qz - pz;
    float e = sqrtf(dx * dx + dy * dy + dz * dz + 1e-20f);
    xc[k][0] = px; xc[k][1] = py; xc[k][2] = pz;
    xc[k][3] = qx; xc[k][4] = qy; xc[k][5] = qz;
    xc[k][6] = dx; xc[k][7] = dy; xc[k][8] = dz; xc[k][9] = e;
  }
  __syncthreads();
  int wave = threadIdx.x >> 5, lane = threadIdx.x & 31;
  int ncol = lane & 15, mb = (lane < 16) ? 0 : 8;
  {                                                // hidden_p: wave covers 64 cols
    v8f acc[4];
    int col0 = wave * 64;
    gemmN<4>(As, CD, W1, CD, col0, CD, b1, lane, acc);
#pragma unroll
    for (int t = 0; t < 4; ++t)
#pragma unroll
      for (int j = 0; j < 8; ++j)
        Hs[(mb + j) * HD + col0 + t * 16 + ncol] = (bf16_t)acc[t][j];
  }
  __syncthreads();
  {                                                // ip + gp + fuse -> feat
    int col0 = wave * 16;
    v8f aip = gemm16(Hs, HD, W2, HD, col0, HD, b2[col0 + ncol], lane);
    v8f agp = gemm16(As, CD, Wgp, CD, col0, CD, bgp[col0 + ncol], lane);
    int c = col0 + ncol;
    float giv = gi[(size_t)bn * CD + c];
    float iiv = ii[(size_t)bn * CD + c];
#pragma unroll
    for (int j = 0; j < 8; ++j) {
      float gp = sigmoidf_(agp[j]);
      float f = (gp * aip[j] + giv * iiv) / (gp + giv);
      feat_out[(size_t)(row0 + mb + j) * CD + c] = f;
    }
  }
  if (threadIdx.x < 128) {                         // enc linear (10 -> 128) + stats0
    int c2 = threadIdx.x;
    float w[10];
#pragma unroll
    for (int i = 0; i < 10; ++i) w[i] = encW[c2 * 10 + i];
    float bb = encB[c2];
    float s = 0.f, s2 = 0.f;
    for (int k = 0; k < 16; ++k) {
      float v = bb;
#pragma unroll
      for (int i = 0; i < 10; ++i) v += w[i] * xc[k][i];
      enclin_out[(size_t)(row0 + k) * CD + c2] = v;
      s += v; s2 += v * v;
    }
    atomicAdd(&stats0[c2], s);
    atomicAdd(&stats0[128 + c2], s2);
  }
}

// ---- mlp2_0: x = [relu(bn0(enclin)), feat] (256) -> y0 (256), stats1 ----
__global__ __launch_bounds__(256) void k_mlp2_0(const float* __restrict__ enclin,
                                                const float* __restrict__ feat,
                                                const float* __restrict__ stats0,
                                                const float* __restrict__ g0, const float* __restrict__ be0,
                                                const bf16_t* __restrict__ W20, const float* __restrict__ b20,
                                                float* __restrict__ y0_out, float* __restrict__ stats1) {
  __shared__ __align__(16) bf16_t Xs[16 * 256];
  int row0 = blockIdx.x * 16;
  const float invR = 1.0f / (float)RD;
  for (int i = threadIdx.x; i < 16 * 256; i += 256) {
    int r = i >> 8, cc = i & 255;
    float v;
    if (cc < 128) {
      float m  = stats0[cc] * invR;
      float vr = stats0[128 + cc] * invR - m * m;
      float x  = enclin[(size_t)(row0 + r) * CD + cc];
      v = fmaxf((x - m) * rsqrtf(vr + 1e-5f) * g0[cc] + be0[cc], 0.0f);
    } else {
      v = feat[(size_t)(row0 + r) * CD + (cc - 128)];
    }
    Xs[i] = (bf16_t)v;
  }
  __syncthreads();
  int wave = threadIdx.x >> 5, lane = threadIdx.x & 31;
  int ncol = lane & 15;
  v8f acc[2];
  int col0 = wave * 32;
  gemmN<2>(Xs, 256, W20, 256, col0, 256, b20, lane, acc);
#pragma unroll
  for (int tt = 0; tt < 2; ++tt) {
    int c0 = col0 + tt * 16;
    store_tile(y0_out, 256, row0, c0, lane, acc[tt]);
    float s = 0.f, s2 = 0.f;
#pragma unroll
    for (int j = 0; j < 8; ++j) { s += acc[tt][j]; s2 += acc[tt][j] * acc[tt][j]; }
    s  += __shfl_xor(s, 16, 32);
    s2 += __shfl_xor(s2, 16, 32);
    if (lane < 16) {
      atomicAdd(&stats1[c0 + lane], s);
      atomicAdd(&stats1[256 + c0 + lane], s2);
    }
  }
}

// ---- mlp2_1: h1 = relu(bn1(y0)) -> y1 (128), stats2 ----
__global__ __launch_bounds__(256) void k_mlp2_1(const float* __restrict__ y0,
                                                const float* __restrict__ stats1,
                                                const float* __restrict__ g1, const float* __restrict__ be1,
                                                const bf16_t* __restrict__ W21, const float* __restrict__ b21,
                                                float* __restrict__ y1_out, float* __restrict__ stats2) {
  __shared__ __align__(16) bf16_t Hs[16 * 256];
  int row0 = blockIdx.x * 16;
  const float invR = 1.0f / (float)RD;
  for (int i = threadIdx.x; i < 16 * 256; i += 256) {
    int r = i >> 8, cc = i & 255;
    float m  = stats1[cc] * invR;
    float vr = stats1[256 + cc] * invR - m * m;
    float x  = y0[(size_t)(row0 + r) * 256 + cc];
    Hs[i] = (bf16_t)fmaxf((x - m) * rsqrtf(vr + 1e-5f) * g1[cc] + be1[cc], 0.0f);
  }
  __syncthreads();
  int wave = threadIdx.x >> 5, lane = threadIdx.x & 31;
  int ncol = lane & 15;
  int col0 = wave * 16;
  v8f acc = gemm16(Hs, 256, W21, 256, col0, 256, b21[col0 + ncol], lane);
  store_tile(y1_out, CD, row0, col0, lane, acc);
  float s = 0.f, s2 = 0.f;
#pragma unroll
  for (int j = 0; j < 8; ++j) { s += acc[j]; s2 += acc[j] * acc[j]; }
  s  += __shfl_xor(s, 16, 32);
  s2 += __shfl_xor(s2, 16, 32);
  if (lane < 16) {
    atomicAdd(&stats2[col0 + lane], s);
    atomicAdd(&stats2[128 + col0 + lane], s2);
  }
}

// ---- final: bn2+relu -> softmax over K -> weighted sum of feat ----
__global__ __launch_bounds__(128) void k_final(const float* __restrict__ y1,
                                               const float* __restrict__ feat,
                                               const float* __restrict__ stats2,
                                               const float* __restrict__ g2, const float* __restrict__ be2,
                                               float* __restrict__ out) {
  int bn = blockIdx.x;
  int c = threadIdx.x;
  const float invR = 1.0f / (float)RD;
  float m  = stats2[c] * invR;
  float vr = stats2[128 + c] * invR - m * m;
  float sc = rsqrtf(vr + 1e-5f) * g2[c];
  float sh = be2[c] - m * sc;
  int row0 = bn * 16;
  float v[16], mx = -3.4e38f;
#pragma unroll
  for (int k = 0; k < 16; ++k) {
    float x = fmaxf(y1[(size_t)(row0 + k) * CD + c] * sc + sh, 0.0f);
    v[k] = x; mx = fmaxf(mx, x);
  }
  float se = 0.f;
#pragma unroll
  for (int k = 0; k < 16; ++k) { v[k] = __expf(v[k] - mx); se += v[k]; }
  float acc = 0.f;
#pragma unroll
  for (int k = 0; k < 16; ++k) acc += v[k] * feat[(size_t)(row0 + k) * CD + c];
  out[(size_t)bn * CD + c] = acc / se;
}

// ---------------- host orchestration ----------------
extern "C" void kernel_launch(void* const* d_in, const int* in_sizes, int n_in,
                              void* d_out, int out_size, void* d_ws, size_t ws_size,
                              hipStream_t stream) {
  const float* wxyz   = (const float*)d_in[0];
  const float* wpts   = (const float*)d_in[1];
  const float* f2xyz  = (const float*)d_in[2];
  const float* f2pts  = (const float*)d_in[3];
  const float* gate_p_W = (const float*)d_in[4];
  const float* gate_p_b = (const float*)d_in[5];
  const float* gate_i_W = (const float*)d_in[6];
  const float* gate_i_b = (const float*)d_in[7];
  const float* inp_p_W1 = (const float*)d_in[8];
  const float* inp_p_b1 = (const float*)d_in[9];
  const float* inp_p_W2 = (const float*)d_in[10];
  const float* inp_p_b2 = (const float*)d_in[11];
  const float* inp_i_W1 = (const float*)d_in[12];
  const float* inp_i_b1 = (const float*)d_in[13];
  const float* inp_i_W2 = (const float*)d_in[14];
  const float* inp_i_b2 = (const float*)d_in[15];
  const float* enc_W  = (const float*)d_in[16];
  const float* enc_b  = (const float*)d_in[17];
  const float* enc_g  = (const float*)d_in[18];
  const float* enc_be = (const float*)d_in[19];
  const float* m20_W  = (const float*)d_in[20];
  const float* m20_b  = (const float*)d_in[21];
  const float* m20_g  = (const float*)d_in[22];
  const float* m20_be = (const float*)d_in[23];
  const float* m21_W  = (const float*)d_in[24];
  const float* m21_b  = (const float*)d_in[25];
  const float* m21_g  = (const float*)d_in[26];
  const float* m21_be = (const float*)d_in[27];

  char* ws = (char*)d_ws;
  int*    idx    = (int*)(ws + OFF_IDX);
  bf16_t* Wb     = (bf16_t*)(ws + OFF_W);
  bf16_t* Xw     = (bf16_t*)(ws + OFF_XW);
  bf16_t* Q      = (bf16_t*)(ws + OFF_Q);
  float*  gi     = (float*)(ws + OFF_GI);
  float*  ii     = (float*)(ws + OFF_II);
  float*  feat   = (float*)(ws + OFF_FEAT);
  float*  enclin = (float*)(ws + OFF_ENCLIN);
  float*  y0     = (float*)(ws + OFF_Y0);
  float*  y1     = (float*)(ws + OFF_Y1);
  float*  stats  = (float*)(ws + OFF_STATS);

  // weight conversions to bf16 (L2-resident for all GEMMs)
  k_cvt4<<<16,  256, 0, stream>>>(gate_p_W, Wb + WGP_O, 16384 / 4);
  k_cvt4<<<16,  256, 0, stream>>>(gate_i_W, Wb + WGI_O, 16384 / 4);
  k_cvt4<<<64,  256, 0, stream>>>(inp_p_W1, Wb + W1P_O, 65536 / 4);
  k_cvt4<<<64,  256, 0, stream>>>(inp_p_W2, Wb + W2P_O, 65536 / 4);
  k_cvt4<<<64,  256, 0, stream>>>(inp_i_W1, Wb + W1I_O, 65536 / 4);
  k_cvt4<<<64,  256, 0, stream>>>(inp_i_W2, Wb + W2I_O, 65536 / 4);
  k_cvt4<<<64,  256, 0, stream>>>(m20_W,    Wb + W20_O, 65536 / 4);
  k_cvt4<<<32,  256, 0, stream>>>(m21_W,    Wb + W21_O, 32768 / 4);
  k_cvt4<<<1024,256, 0, stream>>>(wpts, Xw, (R2D * CD) / 4);
  k_zero<<<4, 256, 0, stream>>>(stats, 1024);

  k_knn<<<dim3(ND / 8, BD), 256, 0, stream>>>(wxyz, f2xyz, idx);
  k_gather<<<(RD * 32) / 256, 256, 0, stream>>>(f2pts, idx, Q);

  k_ibranch<<<R2D / 16, 256, 0, stream>>>(Xw, Wb + WGI_O, gate_i_b,
                                          Wb + W1I_O, inp_i_b1,
                                          Wb + W2I_O, inp_i_b2, gi, ii);

  k_main<<<R2D, 256, 0, stream>>>(Q, Wb + WGP_O, gate_p_b,
                                  Wb + W1P_O, inp_p_b1, Wb + W2P_O, inp_p_b2,
                                  gi, ii, wxyz, f2xyz, idx, enc_W, enc_b,
                                  feat, enclin, stats);

  k_mlp2_0<<<R2D, 256, 0, stream>>>(enclin, feat, stats, enc_g, enc_be,
                                    Wb + W20_O, m20_b, y0, stats + 256);

  k_mlp2_1<<<R2D, 256, 0, stream>>>(y0, stats + 256, m20_g, m20_be,
                                    Wb + W21_O, m21_b, y1, stats + 768);

  k_final<<<R2D, 128, 0, stream>>>(y1, feat, stats + 768, m21_g, m21_be,
                                   (float*)d_out);
}
